// GraphConvolution_60748017434790
// MI455X (gfx1250) — matrix operand order repaired and checked by reading, not compiled, not run on hardware
//
#include <hip/hip_runtime.h>
#include <stdint.h>

// ---------------------------------------------------------------------------
// 3-layer GAT (GATConv + ReLU + BatchNorm, jumping-knowledge concat) on MI455X.
// GEMM path: bf16 WMMA (v_wmma_f32_16x16x32_bf16), f32 accumulation.
// Edge path: wave32-per-edge gather + native f32 L2 atomics (L2-resident: h and
// the accumulator are 51 MB each, both fit in the 192 MB global L2).
// ---------------------------------------------------------------------------

#define NN 100000
#define CC 128
#define NEG_SLOPE 0.2f
#define BN_EPS 1e-5f

typedef __attribute__((ext_vector_type(16))) __bf16 v16bf;
typedef __attribute__((ext_vector_type(8)))  float  v8f;

__device__ __forceinline__ uint16_t f2bf(float f) {
    uint32_t u = __float_as_uint(f);
    // round-to-nearest-even truncation to bf16
    uint32_t r = (u + 0x7FFFu + ((u >> 16) & 1u)) >> 16;
    return (uint16_t)r;
}

__device__ __forceinline__ void atomicMaxFloat(float* addr, float val) {
    // ordered-int trick; addr must be initialized to -inf (0xFF800000)
    if (val >= 0.f) atomicMax((int*)addr, __float_as_int(val));
    else            atomicMin((unsigned int*)addr, __float_as_uint(val));
}

__device__ __forceinline__ void atomAddF32(float* p, float v) {
    __hip_atomic_fetch_add(p, v, __ATOMIC_RELAXED, __HIP_MEMORY_SCOPE_AGENT);
}

// --------------------------- prep kernels ----------------------------------

__global__ void k_f32_to_bf16(const float* __restrict__ in,
                              uint16_t* __restrict__ out, int cnt) {
    int i = blockIdx.x * blockDim.x + threadIdx.x;
    if (i < cnt) out[i] = f2bf(in[i]);
}

// Wt[l][n][k] = W[l][k][n]  (bf16) so consecutive-K pairs are contiguous.
__global__ void k_transpose_w(const float* __restrict__ W,
                              uint16_t* __restrict__ Wt) {
    int i = blockIdx.x * blockDim.x + threadIdx.x;
    if (i >= 3 * CC * CC) return;
    int l = i >> 14;
    int rem = i & 16383;
    int k = rem >> 7;
    int n = rem & 127;
    Wt[(size_t)l * CC * CC + (size_t)n * CC + k] = f2bf(W[i]);
}

__global__ void k_init_layer(float* __restrict__ agg, float* __restrict__ m,
                             float* __restrict__ denom, float* __restrict__ sums,
                             float* __restrict__ sumsq) {
    int i = blockIdx.x * blockDim.x + threadIdx.x;
    if (i < NN * CC) agg[i] = 0.f;
    if (i < NN) { m[i] = __uint_as_float(0xFF800000u); denom[i] = 0.f; }
    if (i < CC) { sums[i] = 0.f; sumsq[i] = 0.f; }
}

// --------------------------- WMMA GEMM -------------------------------------
// H[M,128] = Xb[M,128] (bf16) @ W[128,128], W given pre-transposed bf16 as
// Wt[n*128+k]. One wave per 16x16 output tile, 8 waves/block cover 16 rows x
// 128 cols. K loop: 4 steps of v_wmma_f32_16x16x32_bf16.
__global__ void k_gemm_wmma(const uint16_t* __restrict__ A,   // [M,128] bf16
                            const uint16_t* __restrict__ Bt,  // [128,128] bf16 (n-major)
                            float* __restrict__ H) {
    const int wave = threadIdx.x >> 5;      // 0..7 -> 16-col tile
    const int lane = threadIdx.x & 31;
    const int rowBase = blockIdx.x * 16;
    const int colBase = wave * 16;
    const int half = lane >> 4;             // 0/1
    const int l16  = lane & 15;

    v8f c = {};
    const uint16_t* arow = A  + (size_t)(rowBase + l16) * CC;
    const uint16_t* bcol = Bt + (size_t)(colBase + l16) * CC;

#pragma unroll
    for (int ks = 0; ks < 4; ++ks) {
        const int k0 = ks * 32;
        union { v16bf v; uint32_t u[8]; } a, b;
        // A fragment: lanes 0-15 rows M=l16 carry K = {0..7,16..23}+k0,
        // lanes 16-31 carry K = {8..15,24..31}+k0 (ISA 16-bit A 16x32 layout).
        const int ka = k0 + half * 8;
#pragma unroll
        for (int j = 0; j < 4; ++j) {
            a.u[j]     = *(const uint32_t*)(arow + ka + 2 * j);
            a.u[j + 4] = *(const uint32_t*)(arow + 16 + ka + 2 * j);
        }
        // B fragment: lane col N=l16; lanes 0-15 K = {0..7,16..23}+k0, lanes 16-31
        // K = {8..15,24..31}+k0, two consecutive K per VGPR (matches A layout).
        const int kb = k0 + half * 8;
#pragma unroll
        for (int j = 0; j < 8; ++j) {
            b.u[j] = *(const uint32_t*)(bcol + kb + 2 * j + (j >= 4 ? 8 : 0));
        }
        c = __builtin_amdgcn_wmma_f32_16x16x32_bf16(
                false, a.v, false, b.v, (short)0, c, false, false);
    }

    // D layout: VGPR r, lane L: N = L&15, M = r + 8*(L>>4)
    float* hb = H + (size_t)rowBase * CC + colBase + l16;
#pragma unroll
    for (int r = 0; r < 8; ++r) {
        hb[(size_t)(r + 8 * half) * CC] = c[r];
    }
}

// --------------------------- attention scores ------------------------------
// One wave32 per node: s_src[n] = h[n].a_src ; s_dst[n] = h[n].a_dst
__global__ void k_node_scores(const float* __restrict__ h,
                              const float* __restrict__ a_src,
                              const float* __restrict__ a_dst,
                              float* __restrict__ s_src,
                              float* __restrict__ s_dst) {
    const int wave = threadIdx.x >> 5;
    const int lane = threadIdx.x & 31;
    const int node = blockIdx.x * 8 + wave;
    if (node >= NN) return;
    const float* hr = h + (size_t)node * CC;
    float ss = 0.f, sd = 0.f;
#pragma unroll
    for (int j = 0; j < 4; ++j) {
        float v = hr[lane + 32 * j];
        ss += v * a_src[lane + 32 * j];
        sd += v * a_dst[lane + 32 * j];
    }
#pragma unroll
    for (int off = 16; off; off >>= 1) {
        ss += __shfl_xor(ss, off, 32);
        sd += __shfl_xor(sd, off, 32);
    }
    if (lane == 0) { s_src[node] = ss; s_dst[node] = sd; }
}

// --------------------------- edge kernels ----------------------------------
__device__ __forceinline__ void edge_endpoints(const long long* __restrict__ ei,
                                               long long e, int E, int& src, int& dst) {
    if (e < (long long)E) {
        src = (int)ei[e];
        dst = (int)ei[(long long)E + e];
    } else {
        src = dst = (int)(e - E);
    }
}

__global__ void k_edge_score_max(const long long* __restrict__ ei,
                                 const float* __restrict__ s_src,
                                 const float* __restrict__ s_dst,
                                 float* __restrict__ ebuf,
                                 float* __restrict__ m, int E) {
    long long e = (long long)blockIdx.x * blockDim.x + threadIdx.x;
    if (e >= (long long)E + NN) return;
    int src, dst;
    edge_endpoints(ei, e, E, src, dst);
    float s = s_src[src] + s_dst[dst];
    s = (s > 0.f) ? s : NEG_SLOPE * s;       // leaky relu
    ebuf[e] = s;
    atomicMaxFloat(&m[dst], s);
}

__global__ void k_edge_exp_sum(const long long* __restrict__ ei,
                               float* __restrict__ ebuf,
                               const float* __restrict__ m,
                               float* __restrict__ denom, int E) {
    long long e = (long long)blockIdx.x * blockDim.x + threadIdx.x;
    if (e >= (long long)E + NN) return;
    int src, dst;
    edge_endpoints(ei, e, E, src, dst);
    float ex = __expf(ebuf[e] - m[dst]);
    ebuf[e] = ex;
    atomAddF32(&denom[dst], ex);
}

// One wave32 per edge: agg[dst] += alpha * h[src]; 4 coalesced channel passes.
__global__ void k_edge_aggregate(const long long* __restrict__ ei,
                                 const float* __restrict__ h,
                                 const float* __restrict__ ebuf,
                                 const float* __restrict__ denom,
                                 float* __restrict__ agg, int E) {
    const int wave = threadIdx.x >> 5;
    const int lane = threadIdx.x & 31;
    long long e = (long long)blockIdx.x * 8 + wave;
    if (e >= (long long)E + NN) return;
    int src, dst;
    edge_endpoints(ei, e, E, src, dst);
    float alpha = ebuf[e] / denom[dst];
    const float* hs = h + (size_t)src * CC;
    float* od = agg + (size_t)dst * CC;
#pragma unroll
    for (int j = 0; j < 4; ++j) {
        atomAddF32(&od[lane + 32 * j], alpha * hs[lane + 32 * j]);
    }
}

// --------------------------- bias + relu + BN stats ------------------------
// y = relu(agg + bias) in place; per-channel sum / sumsq via LDS + 1 atomic
// per channel per block.
__global__ void k_bias_relu_stats(float* __restrict__ agg,
                                  const float* __restrict__ bias,
                                  float* __restrict__ sums,
                                  float* __restrict__ sumsq) {
    __shared__ float s1[256], s2[256];
    const int c   = threadIdx.x & 127;
    const int sub = threadIdx.x >> 7;        // 0/1: two rows per iteration
    const float b = bias[c];
    float ls = 0.f, lq = 0.f;
    for (int row = blockIdx.x * 2 + sub; row < NN; row += gridDim.x * 2) {
        size_t idx = (size_t)row * CC + c;
        float v = agg[idx] + b;
        v = (v > 0.f) ? v : 0.f;
        agg[idx] = v;
        ls += v;
        lq += v * v;
    }
    s1[threadIdx.x] = ls;
    s2[threadIdx.x] = lq;
    __syncthreads();
    if (sub == 0) {
        atomAddF32(&sums[c],  s1[c] + s1[c + 128]);
        atomAddF32(&sumsq[c], s2[c] + s2[c + 128]);
    }
}

__global__ void k_bn_params(const float* __restrict__ sums,
                            const float* __restrict__ sumsq,
                            const float* __restrict__ gamma,
                            const float* __restrict__ beta,
                            float* __restrict__ scale,
                            float* __restrict__ shift) {
    int c = threadIdx.x;
    if (c >= CC) return;
    const float invN = 1.0f / (float)NN;
    float mu  = sums[c] * invN;
    float var = sumsq[c] * invN - mu * mu;
    var = (var > 0.f) ? var : 0.f;
    float sc = gamma[c] * __frsqrt_rn(var + BN_EPS);
    scale[c] = sc;
    shift[c] = beta[c] - mu * sc;
}

// v = scale*y + shift -> d_out[:, l*128 : l*128+128] (f32) and xb (bf16, next
// layer GEMM input).
__global__ void k_bn_apply(const float* __restrict__ agg,
                           const float* __restrict__ scale,
                           const float* __restrict__ shift,
                           float* __restrict__ out,
                           uint16_t* __restrict__ xb, int layer) {
    int i = blockIdx.x * blockDim.x + threadIdx.x;
    if (i >= NN * CC) return;
    int n = i >> 7;
    int c = i & 127;
    float v = scale[c] * agg[i] + shift[c];
    out[(size_t)n * (3 * CC) + layer * CC + c] = v;
    xb[i] = f2bf(v);
}

// --------------------------- host orchestration ----------------------------

extern "C" void kernel_launch(void* const* d_in, const int* in_sizes, int n_in,
                              void* d_out, int out_size, void* d_ws, size_t ws_size,
                              hipStream_t stream) {
    const float*     x        = (const float*)d_in[0];
    const long long* ei       = (const long long*)d_in[1];   // int64 [2,E]
    const float*     Ws       = (const float*)d_in[2];       // [3,128,128]
    const float*     att_srcs = (const float*)d_in[3];       // [3,128]
    const float*     att_dsts = (const float*)d_in[4];
    const float*     biases   = (const float*)d_in[5];
    const float*     gammas   = (const float*)d_in[6];
    const float*     betas    = (const float*)d_in[7];
    float*           out      = (float*)d_out;

    const int E  = in_sizes[1] / 2;          // 1,600,000
    const long long ET = (long long)E + NN;  // + self loops

    // ---- workspace carve (512B aligned) ----
    char* p = (char*)d_ws;
    auto carve = [&](size_t bytes) -> void* {
        void* r = (void*)p;
        p += (bytes + 511) & ~(size_t)511;
        return r;
    };
    uint16_t* xb    = (uint16_t*)carve((size_t)NN * CC * 2);   // bf16 activations
    uint16_t* Wt    = (uint16_t*)carve((size_t)3 * CC * CC * 2);
    float*    h     = (float*)carve((size_t)NN * CC * 4);
    float*    agg   = (float*)carve((size_t)NN * CC * 4);
    float*    ssrc  = (float*)carve((size_t)NN * 4);
    float*    sdst  = (float*)carve((size_t)NN * 4);
    float*    m     = (float*)carve((size_t)NN * 4);
    float*    denom = (float*)carve((size_t)NN * 4);
    float*    ebuf  = (float*)carve((size_t)ET * 4);
    float*    sums  = (float*)carve(CC * 4);
    float*    sumsq = (float*)carve(CC * 4);
    float*    scale = (float*)carve(CC * 4);
    float*    shift = (float*)carve(CC * 4);

    const int T = 256;
    const int gNC   = (NN * CC + T - 1) / T;                 // per-element grids
    const int gEdge = (int)((ET + T - 1) / T);               // thread-per-edge
    const int gEw   = (int)((ET + 7) / 8);                   // wave-per-edge
    const int gNode = (NN + 7) / 8;                          // wave-per-node

    // one-time prep
    k_f32_to_bf16<<<gNC, T, 0, stream>>>(x, xb, NN * CC);
    k_transpose_w<<<(3 * CC * CC + T - 1) / T, T, 0, stream>>>(Ws, Wt);

    for (int l = 0; l < 3; ++l) {
        k_init_layer<<<gNC, T, 0, stream>>>(agg, m, denom, sums, sumsq);
        k_gemm_wmma<<<NN / 16, T, 0, stream>>>(xb, Wt + (size_t)l * CC * CC, h);
        k_node_scores<<<gNode, T, 0, stream>>>(h, att_srcs + l * CC,
                                               att_dsts + l * CC, ssrc, sdst);
        k_edge_score_max<<<gEdge, T, 0, stream>>>(ei, ssrc, sdst, ebuf, m, E);
        k_edge_exp_sum<<<gEdge, T, 0, stream>>>(ei, ebuf, m, denom, E);
        k_edge_aggregate<<<gEw, T, 0, stream>>>(ei, h, ebuf, denom, agg, E);
        k_bias_relu_stats<<<1024, T, 0, stream>>>(agg, biases + l * CC, sums, sumsq);
        k_bn_params<<<1, CC, 0, stream>>>(sums, sumsq, gammas + l * CC,
                                          betas + l * CC, scale, shift);
        k_bn_apply<<<gNC, T, 0, stream>>>(agg, scale, shift, out, xb, l);
    }
}
